// AdditiveAttention_79130477461831
// MI455X (gfx1250) — compile-verified, matrix operand order checked by reading, not run on hardware
//
#include <hip/hip_runtime.h>
#include <math.h>
#include <stdint.h>

// ---------------------------------------------------------------------------
// Additive (Bahdanau) attention for MI455X / gfx1250.
//   B=32, S=2048, D=1024, A=1024
//   scores[b,s] = v . tanh(W1 q[b] + W2 k[b,s]);  out = masked softmax over s
// Dominant GEMM (keys x W2^T, 137 GFLOP) on v_wmma_f32_16x16x32_bf16.
// W2 tiles double-buffered into LDS via global_load_async_to_lds_b128
// (ASYNCcnt path), keys staged fp32->bf16 once per workgroup.
// ---------------------------------------------------------------------------

typedef __attribute__((ext_vector_type(16))) __bf16 v16bf;
typedef __attribute__((ext_vector_type(8)))  __bf16 v8bf;
typedef __attribute__((ext_vector_type(4)))  __bf16 v4bf;
typedef __attribute__((ext_vector_type(8)))  float  v8f;

static constexpr int B_    = 32;
static constexpr int S_    = 2048;
static constexpr int D_    = 1024;
static constexpr int A_    = 1024;
static constexpr int LDSPITCH = D_ + 8;          // +8 bf16 pad -> conflict-free ds_read_b128
static constexpr float MASK_NEG = 1e30f;

#if __has_builtin(__builtin_amdgcn_tanhf)
#define FAST_TANH(x) __builtin_amdgcn_tanhf(x)   // gfx1250 V_TANH_F32
#else
#define FAST_TANH(x) tanhf(x)
#endif

// --------------------------- kernel 1: W2 -> bf16 ---------------------------
__global__ void cvt_w2_kernel(const float* __restrict__ W2, __bf16* __restrict__ w2b) {
    int t = blockIdx.x * blockDim.x + threadIdx.x;        // over A_*D_/4 float4s
    const float4 f = ((const float4*)W2)[t];
    v4bf o;
    o[0] = (__bf16)f.x; o[1] = (__bf16)f.y; o[2] = (__bf16)f.z; o[3] = (__bf16)f.w;
    ((v4bf*)w2b)[t] = o;
}

// --------------------------- kernel 2: w1q = q @ W1^T -----------------------
__global__ void w1q_kernel(const float* __restrict__ q, const float* __restrict__ W1,
                           float* __restrict__ w1q) {
    int t = blockIdx.x * blockDim.x + threadIdx.x;        // over B_*A_
    int b = t >> 10, a = t & (A_ - 1);
    const float4* qp = (const float4*)(q  + (size_t)b * D_);
    const float4* wp = (const float4*)(W1 + (size_t)a * D_);
    float s = 0.f;
    for (int i = 0; i < D_ / 4; ++i) {
        float4 qa = qp[i], wa = wp[i];
        s += qa.x * wa.x + qa.y * wa.y + qa.z * wa.z + qa.w * wa.w;
    }
    w1q[t] = s;
}

// --------------------------- kernel 3: fused scores -------------------------
// Grid (S/64, B), 128 threads = 4 waves. Wave w owns s-rows [s0+16w, s0+16w+16).
__global__ __launch_bounds__(128) void
attn_scores_kernel(const float* __restrict__ keys, const __bf16* __restrict__ w2b,
                   const float* __restrict__ w1q,  const float* __restrict__ v,
                   float* __restrict__ out) {
    extern __shared__ char smem_raw[];
    __bf16* sk = (__bf16*)smem_raw;                 // 64 x LDSPITCH  (keys, bf16)
    __bf16* sw = sk + 64 * LDSPITCH;                // 2 x 16 x LDSPITCH (W2 tiles, dbl-buf)

    const int b   = blockIdx.y;
    const int s0  = blockIdx.x * 64;
    const int tid = threadIdx.x;

    // ---- issue async stage of W2 tile 0 into buffer 0 (ASYNCcnt path)
    auto stage_w2_async = [&](int at, int buf) {
        const __bf16* gbase = w2b + (size_t)at * 16 * D_;
        __bf16*       lbase = sw + buf * 16 * LDSPITCH;
        for (int idx = tid; idx < 16 * (D_ / 8); idx += 128) {
            int row = idx >> 7, c = idx & 127;      // D_/8 == 128 x 16B chunks per row
            uint32_t l = (uint32_t)(uintptr_t)(lbase + row * LDSPITCH + c * 8);
            uint64_t g = (uint64_t)(uintptr_t)(gbase + row * D_ + c * 8);
            asm volatile("global_load_async_to_lds_b128 %0, %1, off"
                         :: "v"(l), "v"(g) : "memory");
        }
    };
    stage_w2_async(0, 0);

    // ---- stage 64 key rows: fp32 global -> bf16 LDS (each element read once)
    const float4* kg = (const float4*)(keys + ((size_t)b * S_ + s0) * D_);
    for (int idx = tid; idx < 64 * (D_ / 4); idx += 128) {
        int row = idx >> 8, c = idx & 255;          // D_/4 == 256 chunks per row
        float4 f = kg[row * (D_ / 4) + c];
        v4bf o;
        o[0] = (__bf16)f.x; o[1] = (__bf16)f.y; o[2] = (__bf16)f.z; o[3] = (__bf16)f.w;
        *(v4bf*)(sk + row * LDSPITCH + c * 4) = o;
    }

    const int wv = tid >> 5, lane = tid & 31;
    const int m = lane & 15, hi = lane >> 4;
    // ISA 7.12.2: 16-bit A frag lane(m,hi) holds K = hi*8+{0..7} and 16+hi*8+{0..7}
    const __bf16* aBase = sk + (wv * 16 + m) * LDSPITCH + hi * 8;
    const float* w1qb = w1q + (size_t)b * A_;

    float rowsum[8];
#pragma unroll
    for (int r = 0; r < 8; ++r) rowsum[r] = 0.f;

    for (int at = 0; at < A_ / 16; ++at) {
        // my async loads (tile `at`) must have landed; then publish to all waves
        asm volatile("s_wait_asynccnt 0x0" ::: "memory");
        __syncthreads();

        // stream next W2 tile into the other buffer while we compute
        if (at + 1 < A_ / 16) stage_w2_async(at + 1, (at + 1) & 1);

        const __bf16* bBase = sw + (at & 1) * 16 * LDSPITCH + m * LDSPITCH + hi * 8;

        v8f acc = {};
#pragma unroll 4
        for (int k = 0; k < D_ / 32; ++k) {
            v8bf a0 = *(const v8bf*)(aBase + k * 32);
            v8bf a1 = *(const v8bf*)(aBase + k * 32 + 16);
            v8bf b0 = *(const v8bf*)(bBase + k * 32);
            v8bf b1 = *(const v8bf*)(bBase + k * 32 + 16);
            v16bf af = __builtin_shufflevector(a0, a1, 0,1,2,3,4,5,6,7,8,9,10,11,12,13,14,15);
            v16bf bf = __builtin_shufflevector(b0, b1, 0,1,2,3,4,5,6,7,8,9,10,11,12,13,14,15);
            acc = __builtin_amdgcn_wmma_f32_16x16x32_bf16(
                      false, af, false, bf, (short)0, acc, false, false);
        }
        // epilogue: this lane's N-column is a = at*16 + m for all 8 row-slots
        const int acol = at * 16 + m;
        const float wq = w1qb[acol];
        const float vv = v[acol];
#pragma unroll
        for (int r = 0; r < 8; ++r)
            rowsum[r] += FAST_TANH(acc[r] + wq) * vv;
    }

    // ---- reduce over the 16 lanes of each half-wave (the N dimension)
#pragma unroll
    for (int r = 0; r < 8; ++r) {
        float s = rowsum[r];
        s += __shfl_xor(s, 1, 32);
        s += __shfl_xor(s, 2, 32);
        s += __shfl_xor(s, 4, 32);
        s += __shfl_xor(s, 8, 32);
        rowsum[r] = s;
    }
    if (m == 0) {                                    // lane 0 -> rows M=0..7, lane 16 -> M=8..15
        const int srow = s0 + wv * 16 + hi * 8;
#pragma unroll
        for (int r = 0; r < 8; ++r)
            out[(size_t)b * S_ + srow + r] = rowsum[r];
    }
}

// --------------------------- kernel 4: masked softmax -----------------------
__global__ __launch_bounds__(256) void
softmax_kernel(float* __restrict__ out, const unsigned char* __restrict__ mask) {
    __shared__ float red[256];
    const int b = blockIdx.x, t = threadIdx.x;
    float x[8];
    float mx = -3.0e38f;
#pragma unroll
    for (int i = 0; i < 8; ++i) {
        int s = i * 256 + t;
        float val = out[(size_t)b * S_ + s];
        if (!mask[(size_t)b * S_ + s]) val -= MASK_NEG;
        x[i] = val;
        mx = fmaxf(mx, val);
    }
    red[t] = mx; __syncthreads();
    for (int off = 128; off > 0; off >>= 1) {
        if (t < off) red[t] = fmaxf(red[t], red[t + off]);
        __syncthreads();
    }
    const float gmax = red[0];
    __syncthreads();

    float sum = 0.f;
#pragma unroll
    for (int i = 0; i < 8; ++i) { x[i] = __expf(x[i] - gmax); sum += x[i]; }
    red[t] = sum; __syncthreads();
    for (int off = 128; off > 0; off >>= 1) {
        if (t < off) red[t] += red[t + off];
        __syncthreads();
    }
    const float inv = 1.f / red[0];
#pragma unroll
    for (int i = 0; i < 8; ++i)
        out[(size_t)b * S_ + i * 256 + t] = x[i] * inv;
}

// ---------------------------------------------------------------------------
extern "C" void kernel_launch(void* const* d_in, const int* in_sizes, int n_in,
                              void* d_out, int out_size, void* d_ws, size_t ws_size,
                              hipStream_t stream) {
    const float*         query = (const float*)d_in[0];
    const float*         keys  = (const float*)d_in[1];
    const unsigned char* mask  = (const unsigned char*)d_in[2];  // jax bool -> 1 byte
    const float*         W1    = (const float*)d_in[3];
    const float*         W2    = (const float*)d_in[4];
    const float*         v     = (const float*)d_in[5];
    float*               out   = (float*)d_out;

    // workspace layout: [0, 2MB) W2 as bf16; then w1q (B_*A_ fp32, 128KB)
    __bf16* w2b = (__bf16*)d_ws;
    float*  w1q = (float*)((char*)d_ws + (size_t)A_ * D_ * sizeof(__bf16));

    cvt_w2_kernel<<<(A_ * D_ / 4) / 256, 256, 0, stream>>>(W2, w2b);
    w1q_kernel<<<(B_ * A_) / 256, 256, 0, stream>>>(query, W1, w1q);

    const size_t lds_bytes = (size_t)(64 + 32) * LDSPITCH * sizeof(__bf16); // ~194 KB < 320 KB
    attn_scores_kernel<<<dim3(S_ / 64, B_), 128, lds_bytes, stream>>>(keys, w2b, w1q, v, out);

    softmax_kernel<<<B_, 256, 0, stream>>>(out, mask);
}
